// HGATLayer_34196529611343
// MI455X (gfx1250) — matrix operand order repaired and checked by reading, not compiled, not run on hardware
//
#include <hip/hip_runtime.h>

#define EPS_C      1e-7f
#define MIN_NORM_C 1e-15f
#define PROJ_EPS_C 0.004f
#define NEG_SLOPE_C 0.2f

typedef __attribute__((ext_vector_type(16))) __bf16 v16bf;
typedef __attribute__((ext_vector_type(8)))  float  v8f;
typedef __attribute__((ext_vector_type(4)))  unsigned v4u;
typedef __attribute__((ext_vector_type(8)))  int  v8i;
typedef __attribute__((ext_vector_type(4)))  int  v4i;

#if defined(__has_builtin)
#if __has_builtin(__builtin_amdgcn_tensor_load_to_lds)
#define HGAT_TDM 1
#endif
#endif
#ifndef HGAT_TDM
#define HGAT_TDM 0
#endif

union FragBF { v16bf v; unsigned u[8]; };

// ---------- small device helpers ----------
__device__ __forceinline__ unsigned short f2bf(float f) {
    unsigned u = __float_as_uint(f);
    u += 0x7FFFu + ((u >> 16) & 1u);   // round-to-nearest-even
    return (unsigned short)(u >> 16);
}
__device__ __forceinline__ float wsum32(float v) {
#pragma unroll
    for (int m = 16; m >= 1; m >>= 1) v += __shfl_xor(v, m, 32);
    return v;
}
__device__ __forceinline__ float gsum8(float v) {
#pragma unroll
    for (int m = 4; m >= 1; m >>= 1) v += __shfl_xor(v, m, 8);
    return v;
}
__device__ __forceinline__ float dot4(float4 a, float4 b) {
    return a.x * b.x + a.y * b.y + a.z * b.z + a.w * b.w;
}
__device__ __forceinline__ float4 f4scale(float4 a, float s) {
    return make_float4(a.x * s, a.y * s, a.z * s, a.w * s);
}
__device__ __forceinline__ float4 f4axpby(float a, float4 x, float b, float4 y) {
    return make_float4(a * x.x + b * y.x, a * x.y + b * y.y,
                       a * x.z + b * y.z, a * x.w + b * y.w);
}
__device__ __forceinline__ float artanh_c(float v) {
    v = fminf(fmaxf(v, -1.f + EPS_C), 1.f - EPS_C);
    return 0.5f * (log1pf(v) - log1pf(-v));
}
// monotone float<->uint key for atomicMax on signed floats
__device__ __forceinline__ unsigned fkey(float f) {
    unsigned u = __float_as_uint(f);
    return (u & 0x80000000u) ? ~u : (u | 0x80000000u);
}
__device__ __forceinline__ float funkey(unsigned k) {
    unsigned u = (k & 0x80000000u) ? (k & 0x7FFFFFFFu) : ~k;
    return __uint_as_float(u);
}

// ---------- init / conversion kernels ----------
__global__ void hgat_zero_f32(float* p, int n) {
    int i = blockIdx.x * blockDim.x + threadIdx.x;
    if (i < n) p[i] = 0.f;
}
__global__ void hgat_zero_u32(unsigned* p, int n) {
    int i = blockIdx.x * blockDim.x + threadIdx.x;
    if (i < n) p[i] = 0u;   // key(very negative) sentinel; every dst has a self loop
}
__global__ void hgat_w_bf16(const float* __restrict__ W, unsigned short* __restrict__ wbf, int n) {
    int i = blockIdx.x * blockDim.x + threadIdx.x;
    if (i < n) wbf[i] = f2bf(W[i]);
}

// ---------- hyp_bias = proj(expmap0(lin_bias)) (one wave) ----------
__global__ void hgat_hyp_bias(const float* __restrict__ lb, float* __restrict__ hb) {
    int lane = threadIdx.x & 31;
    float4 u = ((const float4*)lb)[lane];
    float n2 = wsum32(dot4(u, u));
    float n  = fmaxf(sqrtf(n2), MIN_NORM_C);
    float4 b = f4scale(u, tanhf(n) / n);          // sqrt(C)=1
    float bn2 = wsum32(dot4(b, b));
    float bn  = fmaxf(sqrtf(bn2), MIN_NORM_C);
    float maxn = 1.f - PROJ_EPS_C;
    if (bn > maxn) b = f4scale(b, maxn / bn);
    ((float4*)hb)[lane] = b;
}

// ---------- WMMA GEMM: mx = x @ W^T  (bf16 in, f32 accumulate) ----------
// W (bf16, 32KB) staged to LDS via the Tensor Data Mover when available.
__global__ void __launch_bounds__(256)
hgat_gemm_wmma(const float* __restrict__ x, const unsigned short* __restrict__ wbf,
               float* __restrict__ mx, int N) {
    __shared__ unsigned short Wlds[128 * 128];    // 32 KB bf16 copy of W, shared by 8 waves
    const int tid = threadIdx.x;
    const int wid = tid >> 5, lane = tid & 31;

#if HGAT_TDM
    if (wid == 0) {
        // Tensor DMA descriptor (D#): 1 valid descriptor, type=2 ("image"),
        // data_size=2B, tensor/tile = one 16384-element row -> 32KB into LDS.
        unsigned long long ga = (unsigned long long)wbf;
        unsigned ldsoff = (unsigned)(unsigned long long)(void*)&Wlds[0];
        v4u g0 = {1u,                                   // count=1, user descriptor
                  ldsoff,                               // lds_addr
                  (unsigned)ga,                         // global_addr[31:0]
                  (unsigned)(((ga >> 32) & 0x1FFFFFFu) | (2u << 30))}; // [56:32] | type=2
        v8i g1 = {(int)(1u << 16),        // workgroup_mask=0, data_size=1 (2 bytes)
                  (int)(0x4000u << 16),   // tensor_dim0[15:0]=16384 @ bits[63:48]
                  (int)(1u << 16),        // tensor_dim0[31:16]=0, tensor_dim1=1
                  (int)(0x4000u << 16),   // tensor_dim1[31:16]=0, tile_dim0=16384
                  1,                      // tile_dim1=1, tile_dim2=0
                  16384,                  // tensor_dim0_stride[31:0]
                  (int)(0x4000u << 16),   // stride0[47:32]=0, tensor_dim1_stride[15:0]
                  0};
        v4i g2 = {0, 0, 0, 0};
        v4i g3 = {0, 0, 0, 0};
        v8i g4 = {0, 0, 0, 0, 0, 0, 0, 0};
        __builtin_amdgcn_tensor_load_to_lds(g0, g1, g2, g3, g4, 0);
        __builtin_amdgcn_s_wait_tensorcnt(0);
    }
    __syncthreads();
    // The TDM wrote LDS behind the compiler's back (descriptor, not pointer):
    // force the compiler to treat Wlds as written so B-fragment loads survive.
    asm volatile("" ::: "memory");
#else
    for (int i = tid; i < 128 * 128; i += 256) Wlds[i] = wbf[i];
    __syncthreads();
#endif

    const int m = lane & 15, half = lane >> 4;
    const int rowbase = blockIdx.x * 128 + wid * 16;
    const int arow = rowbase + m;
    // Clamp instead of masking: D row m depends only on A row m, and
    // out-of-range D rows are never stored, so garbage there is harmless.
    const float* xrow = x + (size_t)((arow < N) ? arow : (N - 1)) * 128;

    // A fragments: 16x32 bf16 per K-step.
    FragBF fa[4];
#pragma unroll
    for (int ks = 0; ks < 4; ++ks) {
#pragma unroll
        for (int i = 0; i < 8; ++i) {
            int k = ks * 32 + ((i >= 4) ? 16 : 0) + half * 8 + (i & 3) * 2;
            float2 p = *(const float2*)(xrow + k);
            fa[ks].u[i] = (unsigned)f2bf(p.x) | ((unsigned)f2bf(p.y) << 16);
        }
    }

    const int n = m;  // output column within 16-wide tile
    const bool full = (rowbase + 16 <= N);   // wave-uniform fast path
#pragma unroll
    for (int ct = 0; ct < 8; ++ct) {
        v8f acc = {0.f, 0.f, 0.f, 0.f, 0.f, 0.f, 0.f, 0.f};
#pragma unroll
        for (int ks = 0; ks < 4; ++ks) {
            // B = W^T tile (32x16): B[k, nc] = W[ct*16+nc, k]
            FragBF fb;
            const unsigned short* wrow = &Wlds[(ct * 16 + n) * 128 + ks * 32 + half * 8];
#pragma unroll
            for (int i = 0; i < 8; ++i)
                fb.u[i] = *(const unsigned*)(wrow + (i >> 2) * 16 + (i & 3) * 2);
            acc = __builtin_amdgcn_wmma_f32_16x16x32_bf16(
                false, fa[ks].v, false, fb.v, (short)0, acc, false, false);
        }
        float* orow = mx + (size_t)(rowbase + 8 * half) * 128 + ct * 16 + n;
        if (full) {
#pragma unroll
            for (int v = 0; v < 8; ++v) orow[(size_t)v * 128] = acc[v];
        } else {
#pragma unroll
            for (int v = 0; v < 8; ++v)
                if (rowbase + 8 * half + v < N) orow[(size_t)v * 128] = acc[v];
        }
    }
}

// ---------- per-row hyperbolic transform -> log_x, s_i, s_j (one wave per row) ----------
__global__ void __launch_bounds__(256)
hgat_row_transform(const float* __restrict__ x, const float* __restrict__ mx,
                   const float* __restrict__ hb, const float* __restrict__ att,
                   float* __restrict__ lxo, float* __restrict__ si,
                   float* __restrict__ sj, int N) {
    const int wid = threadIdx.x >> 5, lane = threadIdx.x & 31;
    const int row = blockIdx.x * 8 + wid;
    if (row >= N) return;
    const float maxn = 1.f - PROJ_EPS_C;   // sqrt(C)=1

    float4 xv = ((const float4*)(x + (size_t)row * 128))[lane];
    float4 mv = ((const float4*)(mx + (size_t)row * 128))[lane];

    // mobius_matvec tail
    float xn2 = wsum32(dot4(xv, xv));
    float mxn2 = wsum32(dot4(mv, mv));
    float xn  = fmaxf(sqrtf(xn2), MIN_NORM_C);
    float mxn = fmaxf(sqrtf(mxn2), MIN_NORM_C);
    float t   = tanhf((mxn / xn) * artanh_c(xn));
    float fac = (mxn2 == 0.f) ? 0.f : (t / mxn);
    float4 h = f4scale(mv, fac);
    // proj
    float rn2 = wsum32(dot4(h, h));
    float rn  = fmaxf(sqrtf(rn2), MIN_NORM_C);
    if (rn > maxn) h = f4scale(h, maxn / rn);
    float x2 = fminf(rn, maxn); x2 *= x2;

    // mobius_add(h, hyp_bias) (C=1)
    float4 b = ((const float4*)hb)[lane];
    float y2 = wsum32(dot4(b, b));
    float xy = wsum32(dot4(h, b));
    float cA = 1.f + 2.f * xy + y2;
    float cB = 1.f - x2;
    float den = fmaxf(1.f + 2.f * xy + x2 * y2, MIN_NORM_C);
    float4 h2 = f4scale(f4axpby(cA, h, cB, b), 1.f / den);
    // proj
    float n2 = wsum32(dot4(h2, h2));
    float nn = fmaxf(sqrtf(n2), MIN_NORM_C);
    if (nn > maxn) h2 = f4scale(h2, maxn / nn);
    float np = fminf(nn, maxn);

    // logmap0
    float4 lx = f4scale(h2, artanh_c(np) / np);
    ((float4*)(lxo + (size_t)row * 128))[lane] = lx;

    // per-head attention dots: head = lane/8, within-head dims = (lane%8)*4 ..
    int hh = lane >> 3;
    int w  = (lane & 7) * 4;
    const float* ai = att + hh * 64 + w;        // att_i = att[h, 0:32]
    const float* aj = att + hh * 64 + 32 + w;   // att_j = att[h, 32:64]
    float pi = lx.x * ai[0] + lx.y * ai[1] + lx.z * ai[2] + lx.w * ai[3];
    float pj = lx.x * aj[0] + lx.y * aj[1] + lx.z * aj[2] + lx.w * aj[3];
    pi = gsum8(pi);
    pj = gsum8(pj);
    if ((lane & 7) == 0) {
        si[row * 4 + hh] = pi;
        sj[row * 4 + hh] = pj;
    }
}

// ---------- edge pass 1: raw scores + segment max (keyed atomicMax) ----------
__global__ void hgat_edge_score_max(const int* __restrict__ esrc, const int* __restrict__ edst,
                                    const float* __restrict__ si, const float* __restrict__ sj,
                                    float* __restrict__ ex, unsigned* __restrict__ mk,
                                    int E, int N) {
    int e = blockIdx.x * blockDim.x + threadIdx.x;
    if (e >= E + N) return;
    int s = (e < E) ? esrc[e] : (e - E);
    int d = (e < E) ? edst[e] : (e - E);
    float4 a = *(const float4*)(si + 4 * (size_t)d);
    float4 b = *(const float4*)(sj + 4 * (size_t)s);
    float v[4] = {a.x + b.x, a.y + b.y, a.z + b.z, a.w + b.w};
#pragma unroll
    for (int h = 0; h < 4; ++h) {
        float sv = (v[h] > 0.f) ? v[h] : NEG_SLOPE_C * v[h];  // leaky_relu
        ex[(size_t)e * 4 + h] = sv;
        atomicMax(&mk[d * 4 + h], fkey(sv));
    }
}

// ---------- edge pass 2: exp(score - max), segment sum ----------
__global__ void hgat_edge_exp_sum(const int* __restrict__ edst,
                                  float* __restrict__ ex, const unsigned* __restrict__ mk,
                                  float* __restrict__ den, int E, int N) {
    int e = blockIdx.x * blockDim.x + threadIdx.x;
    if (e >= E + N) return;
    int d = (e < E) ? edst[e] : (e - E);
#pragma unroll
    for (int h = 0; h < 4; ++h) {
        float mv = funkey(mk[d * 4 + h]);
        float ev = __expf(ex[(size_t)e * 4 + h] - mv);
        ex[(size_t)e * 4 + h] = ev;
        atomicAdd(&den[d * 4 + h], ev);
    }
}

// ---------- edge pass 3: out[dst] += log_x[src] * alpha  (one wave per edge) ----------
__global__ void __launch_bounds__(256)
hgat_edge_scatter(const int* __restrict__ esrc, const int* __restrict__ edst,
                  const float* __restrict__ lx, const float* __restrict__ ex,
                  const float* __restrict__ den, float* __restrict__ out,
                  int E, int N) {
    int gw = (blockIdx.x * blockDim.x + threadIdx.x) >> 5;
    int lane = threadIdx.x & 31;
    if (gw >= E + N) return;
    int s = (gw < E) ? esrc[gw] : (gw - E);
    int d = (gw < E) ? edst[gw] : (gw - E);
    int h = lane >> 3;
    float w = ex[(size_t)gw * 4 + h] / fmaxf(den[d * 4 + h], MIN_NORM_C);
    float4 v = ((const float4*)(lx + (size_t)s * 128))[lane];
    float* o = out + (size_t)d * 128 + lane * 4;
    atomicAdd(o + 0, v.x * w);
    atomicAdd(o + 1, v.y * w);
    atomicAdd(o + 2, v.z * w);
    atomicAdd(o + 3, v.w * w);
}

// ---------- finalize: relu(acc + conv_bias) -> proj(expmap0(.)) ----------
__global__ void __launch_bounds__(256)
hgat_finalize(float* __restrict__ out, const float* __restrict__ cb, int N) {
    const int wid = threadIdx.x >> 5, lane = threadIdx.x & 31;
    const int row = blockIdx.x * 8 + wid;
    if (row >= N) return;
    float4 u = ((const float4*)(out + (size_t)row * 128))[lane];
    float4 b = ((const float4*)cb)[lane];
    u.x = fmaxf(u.x + b.x, 0.f);
    u.y = fmaxf(u.y + b.y, 0.f);
    u.z = fmaxf(u.z + b.z, 0.f);
    u.w = fmaxf(u.w + b.w, 0.f);
    float n2 = wsum32(dot4(u, u));
    float n  = fmaxf(sqrtf(n2), MIN_NORM_C);
    float4 r = f4scale(u, tanhf(n) / n);          // expmap0, sqrt(C)=1
    float rn2 = wsum32(dot4(r, r));
    float rn  = fmaxf(sqrtf(rn2), MIN_NORM_C);
    float maxn = 1.f - PROJ_EPS_C;
    if (rn > maxn) r = f4scale(r, maxn / rn);
    ((float4*)(out + (size_t)row * 128))[lane] = r;
}

// ---------- host launch ----------
extern "C" void kernel_launch(void* const* d_in, const int* in_sizes, int n_in,
                              void* d_out, int out_size, void* d_ws, size_t ws_size,
                              hipStream_t stream) {
    const float* x   = (const float*)d_in[0];
    const float* W   = (const float*)d_in[1];
    const float* lb  = (const float*)d_in[2];
    const float* att = (const float*)d_in[3];
    const float* cb  = (const float*)d_in[4];
    const int* esrc  = (const int*)d_in[5];
    const int* edst  = (const int*)d_in[6];
    const int N  = in_sizes[0] / 128;
    const int E  = in_sizes[5];
    const int ET = E + N;
    float* out = (float*)d_out;
    (void)n_in; (void)out_size; (void)ws_size;

    char* ws = (char*)d_ws;
    size_t off = 0;
    auto carve = [&](size_t bytes) -> char* {
        char* p = ws + off;
        off += (bytes + 255) & ~(size_t)255;
        return p;
    };
    float*          mx  = (float*)carve((size_t)N * 128 * sizeof(float));
    float*          lx  = (float*)carve((size_t)N * 128 * sizeof(float));
    float*          si  = (float*)carve((size_t)N * 4 * sizeof(float));
    float*          sj  = (float*)carve((size_t)N * 4 * sizeof(float));
    unsigned*       mk  = (unsigned*)carve((size_t)N * 4 * sizeof(unsigned));
    float*          den = (float*)carve((size_t)N * 4 * sizeof(float));
    float*          ex  = (float*)carve((size_t)ET * 4 * sizeof(float));
    float*          hb  = (float*)carve(128 * sizeof(float));
    unsigned short* wbf = (unsigned short*)carve(16384 * sizeof(unsigned short));

    hgat_zero_f32<<<dim3((N * 128 + 255) / 256), dim3(256), 0, stream>>>(out, N * 128);
    hgat_zero_u32<<<dim3((N * 4 + 255) / 256), dim3(256), 0, stream>>>(mk, N * 4);
    hgat_zero_f32<<<dim3((N * 4 + 255) / 256), dim3(256), 0, stream>>>(den, N * 4);
    hgat_w_bf16<<<dim3(64), dim3(256), 0, stream>>>(W, wbf, 16384);
    hgat_hyp_bias<<<dim3(1), dim3(32), 0, stream>>>(lb, hb);
    hgat_gemm_wmma<<<dim3((N + 127) / 128), dim3(256), 0, stream>>>(x, wbf, mx, N);
    hgat_row_transform<<<dim3((N + 7) / 8), dim3(256), 0, stream>>>(x, mx, hb, att, lx, si, sj, N);
    hgat_edge_score_max<<<dim3((ET + 255) / 256), dim3(256), 0, stream>>>(esrc, edst, si, sj, ex, mk, E, N);
    hgat_edge_exp_sum<<<dim3((ET + 255) / 256), dim3(256), 0, stream>>>(edst, ex, mk, den, E, N);
    hgat_edge_scatter<<<dim3((ET + 7) / 8), dim3(256), 0, stream>>>(esrc, edst, lx, ex, den, out, E, N);
    hgat_finalize<<<dim3((N + 7) / 8), dim3(256), 0, stream>>>(out, cb, N);
}
